// GNNBasisDiscoverer_49409303773444
// MI455X (gfx1250) — matrix operand order repaired and checked by reading, not compiled, not run on hardware
//
#include <hip/hip_runtime.h>
#include <math.h>

typedef __attribute__((ext_vector_type(16))) _Float16 v16h;
typedef __attribute__((ext_vector_type(8)))  float    v8f;

namespace {

constexpr int   kN   = 128;      // nodes per graph
constexpr int   kD   = 64;       // model dim
constexpr int   kH   = 2;        // heads
constexpr int   kDH  = 32;       // head dim
constexpr int   kL   = 2;        // layers
constexpr int   kK   = 8;        // top-k neighbors
constexpr int   kNB  = 6;        // basis functions
constexpr int   kM   = 8 * 256;  // B*T independent graphs
constexpr float kEps = 1e-5f;

// ~129.5 KB of LDS: entire per-graph state stays on-chip.
struct __align__(32) Smem {
  float    h[kN][kD];         // 32 KB persistent node features (f32)
  _Float16 qkv[3][kN][kD];    // 48 KB; qkv[0] doubles as hidden / attn-out A operand
  _Float16 w0[kD][kD];        // 8 KB  weight slot A (f16)
  _Float16 w1[kD][kD];        // 8 KB  weight slot B (f16)
  float    scores[64][kN];    // 32 KB row-block of attention scores (also f32 staging)
  float    bias[kD];
  float    bias2[kD];
  float    lng[kD];
  float    lnb[kD];
  float    xrow[kN];          // async-loaded x row for this graph
};

__device__ inline v8f wmma16(v16h a, v16h b, v8f c) {
  // D = A(16x32 f16) * B(32x16 f16) + C(16x16 f32)
  return __builtin_amdgcn_wmma_f32_16x16x32_f16(false, a, false, b, (short)0, c,
                                                false, false);
}

// A operand (16x32 tile of a row-major f16 matrix, leading dim ld).
// Lane L holds row row0+(L&15); halves 0..7 = K kb..kb+7, halves 8..15 = K kb+16..kb+23,
// with kb = k0 + 8*(L>>4)  (per CDNA5 ISA 16-bit A layout).
__device__ inline v16h load_a_f16(const _Float16* S, int ld, int row0, int k0) {
  const int lane = threadIdx.x & 31;
  const _Float16* p = S + (row0 + (lane & 15)) * ld + k0 + ((lane >> 4) << 3);
  v16h a;
#pragma unroll
  for (int j = 0; j < 8; ++j) { a[j] = p[j]; a[j + 8] = p[j + 16]; }
  return a;
}

// Same, but converting from an f32 source on the fly.
__device__ inline v16h load_a_f32(const float* S, int ld, int row0, int k0) {
  const int lane = threadIdx.x & 31;
  const float* p = S + (row0 + (lane & 15)) * ld + k0 + ((lane >> 4) << 3);
  v16h a;
#pragma unroll
  for (int j = 0; j < 8; ++j) { a[j] = (_Float16)p[j]; a[j + 8] = (_Float16)p[j + 16]; }
  return a;
}

// B operand for C = A @ W^T: B(K=k, N=n) = W[n][k], W row-major (ld).
// Lane L holds column n0+(L&15); halves 0..7 = K kb..kb+7, halves 8..15 = K kb+16..kb+23,
// with kb = k0 + 8*(L>>4) (matches A layout).
__device__ inline v16h load_bt(const _Float16* W, int ld, int n0, int k0) {
  const int lane = threadIdx.x & 31;
  const _Float16* p = W + (n0 + (lane & 15)) * ld + k0 + ((lane >> 4) << 3);
  v16h b;
#pragma unroll
  for (int j = 0; j < 8; ++j) { b[j] = p[j]; b[j + 8] = p[j + 16]; }
  return b;
}

// One 16x16 f32 tile of A(128xkD) @ W^T(kDxkD), K = 64 (two WMMAs).
__device__ inline v8f mm_tile_hf32(const float* A, const _Float16* W, int row0, int col0) {
  v8f acc = {};
  acc = wmma16(load_a_f32(A, kD, row0, 0),  load_bt(W, kD, col0, 0),  acc);
  acc = wmma16(load_a_f32(A, kD, row0, 32), load_bt(W, kD, col0, 32), acc);
  return acc;
}
__device__ inline v8f mm_tile_hf16(const _Float16* A, const _Float16* W, int row0, int col0) {
  v8f acc = {};
  acc = wmma16(load_a_f16(A, kD, row0, 0),  load_bt(W, kD, col0, 0),  acc);
  acc = wmma16(load_a_f16(A, kD, row0, 32), load_bt(W, kD, col0, 32), acc);
  return acc;
}

__device__ inline float gelu_exact(float v) {
  return 0.5f * v * (1.0f + erff(v * 0.70710678118654752f));
}

__global__ __launch_bounds__(256, 1)
void gnn_basis_kernel(const float* __restrict__ x,
                      const float* __restrict__ w_in1, const float* __restrict__ b_in1,
                      const float* __restrict__ w_in2, const float* __restrict__ b_in2,
                      const float* __restrict__ species,
                      const float* __restrict__ wq, const float* __restrict__ bq,
                      const float* __restrict__ wk, const float* __restrict__ bk,
                      const float* __restrict__ wvw, const float* __restrict__ bv,
                      const float* __restrict__ wo, const float* __restrict__ bo,
                      const float* __restrict__ lngp, const float* __restrict__ lnbp,
                      const float* __restrict__ w_basis, const float* __restrict__ b_basis,
                      float* __restrict__ out) {
  __shared__ Smem s;
  const int m    = blockIdx.x;     // graph row in [0, B*T)
  const int tid  = threadIdx.x;    // 256 threads = 8 wave32
  const int wid  = tid >> 5;
  const int lane = tid & 31;

  float* stage = &s.scores[0][0];  // f32 staging region

  // ---- Async DMA of this graph's x row into LDS (CDNA5 ASYNCcnt path),
  //      overlapped with weight staging below. Each lane copies its own element;
  //      generic->LDS address uses the ISA's addr[31:0] truncation rule.
  if (tid < kN) {
    const unsigned ldsdst = (unsigned)(uintptr_t)&s.xrow[tid];
    const float*   gsrc   = x + m * kN + tid;
    asm volatile("global_load_async_to_lds_b32 %0, %1, off"
                 :: "v"(ldsdst), "v"(gsrc)
                 : "memory");
  }

  // ================= Embed: feats -> Linear(3,64) -> gelu -> Linear(64,64)+species
  // stage w_in1 (64x3) + b_in1, w_in2 as f16, b_in2.
  if (tid < 192)                stage[tid] = w_in1[tid];
  if (tid >= 192 && tid < 256)  stage[tid] = b_in1[tid - 192];
  for (int i = tid; i < kD * kD; i += 256) (&s.w0[0][0])[i] = (_Float16)w_in2[i];
  if (tid < kD) s.bias2[tid] = b_in2[tid];
  asm volatile("s_wait_asynccnt 0" ::: "memory");
  __syncthreads();

  if (tid < kN) {
    const float xv = s.xrow[tid];
    const float sf = fmaxf(xv, 1e-6f);
    const float f0 = xv, f1 = logf(sf), f2 = sqrtf(sf);
#pragma unroll 4
    for (int j = 0; j < kD; ++j) {
      const float hv = f0 * stage[j * 3 + 0] + f1 * stage[j * 3 + 1] +
                       f2 * stage[j * 3 + 2] + stage[192 + j];
      s.qkv[0][tid][j] = (_Float16)gelu_exact(hv);   // hidden activations (A operand)
    }
  }
  __syncthreads();

  {  // h = hidden @ w_in2^T + b_in2 + species_emb   (WMMA, f32 out)
    const int row0 = wid * 16;
    for (int ct = 0; ct < 4; ++ct) {
      const int col0 = ct * 16;
      const v8f acc = mm_tile_hf16(&s.qkv[0][0][0], &s.w0[0][0], row0, col0);
      const int col = col0 + (lane & 15);
#pragma unroll
      for (int r = 0; r < 8; ++r) {
        const int row = row0 + r + ((lane >> 4) << 3);
        s.h[row][col] = acc[r] + s.bias2[col] + species[row * kD + col];
      }
    }
  }
  __syncthreads();

  // ================= Sparse-GAT layers
  for (int l = 0; l < kL; ++l) {
    const float* Wq = wq  + l * kD * kD;
    const float* Wk = wk  + l * kD * kD;
    const float* Wv = wvw + l * kD * kD;
    const float* Wo = wo  + l * kD * kD;
    __builtin_prefetch(Wv, 0, 1);               // global_prefetch_b8
    __builtin_prefetch(Wo, 0, 1);

    // ---- phase 1: Wq,Wk resident; project q,k
    for (int i = tid; i < kD * kD; i += 256) {
      (&s.w0[0][0])[i] = (_Float16)Wq[i];
      (&s.w1[0][0])[i] = (_Float16)Wk[i];
    }
    if (tid < kD) { s.bias[tid] = bq[l * kD + tid]; s.bias2[tid] = bk[l * kD + tid]; }
    __syncthreads();
    {
      const int row0 = wid * 16;
      for (int ct = 0; ct < 4; ++ct) {
        const int col0 = ct * 16;
        const v8f aq = mm_tile_hf32(&s.h[0][0], &s.w0[0][0], row0, col0);
        const v8f ak = mm_tile_hf32(&s.h[0][0], &s.w1[0][0], row0, col0);
        const int col = col0 + (lane & 15);
#pragma unroll
        for (int r = 0; r < 8; ++r) {
          const int row = row0 + r + ((lane >> 4) << 3);
          s.qkv[0][row][col] = (_Float16)(aq[r] + s.bias[col]);
          s.qkv[1][row][col] = (_Float16)(ak[r] + s.bias2[col]);
        }
      }
    }
    __syncthreads();

    // ---- phase 2: Wv,Wo resident; project v; stage LN params
    for (int i = tid; i < kD * kD; i += 256) {
      (&s.w0[0][0])[i] = (_Float16)Wv[i];
      (&s.w1[0][0])[i] = (_Float16)Wo[i];
    }
    if (tid < kD) { s.bias[tid] = bv[l * kD + tid]; s.bias2[tid] = bo[l * kD + tid]; }
    if (tid >= 64 && tid < 128) {
      s.lng[tid - 64] = lngp[l * kD + tid - 64];
      s.lnb[tid - 64] = lnbp[l * kD + tid - 64];
    }
    __syncthreads();
    {
      const int row0 = wid * 16;
      for (int ct = 0; ct < 4; ++ct) {
        const int col0 = ct * 16;
        const v8f av = mm_tile_hf32(&s.h[0][0], &s.w0[0][0], row0, col0);
        const int col = col0 + (lane & 15);
#pragma unroll
        for (int r = 0; r < 8; ++r) {
          const int row = row0 + r + ((lane >> 4) << 3);
          s.qkv[2][row][col] = (_Float16)(av[r] + s.bias[col]);
        }
      }
    }
    __syncthreads();

    // ---- attention: per head, per 64-row block; scores via WMMA (K = DH = 32)
    for (int head = 0; head < kH; ++head) {
      for (int rb = 0; rb < 2; ++rb) {
        const int rowbase = rb * 64;
        const int rt = wid >> 1;                         // row tile 0..3 within block
        const v16h aq = load_a_f16(&s.qkv[0][0][0], kD, rowbase + rt * 16, head * kDH);
        for (int i = 0; i < 4; ++i) {
          const int col0 = ((wid & 1) * 4 + i) * 16;     // column tiles 0..7
          const v16h bk16 = load_bt(&s.qkv[1][0][0], kD, col0, head * kDH);
          v8f acc = {};
          acc = wmma16(aq, bk16, acc);
          const int col = col0 + (lane & 15);
#pragma unroll
          for (int r = 0; r < 8; ++r)
            s.scores[rt * 16 + r + ((lane >> 4) << 3)][col] =
                acc[r] * 0.17677669529663687f;           // * DH^-0.5
        }
        __syncthreads();

        // top-8 + softmax + sparse attn@V: one thread per node row.
        if (tid < 64) {
          const int grow = rowbase + tid;
          float tvv[kK]; int tii[kK];
#pragma unroll
          for (int i = 0; i < kK; ++i) { tvv[i] = -3.0e38f; tii[i] = 0; }
          for (int c = 0; c < kN; ++c) {
            float sc = s.scores[tid][c];
            int   ci = c;
#pragma unroll
            for (int q2 = 0; q2 < kK; ++q2) {           // branch-free insertion net
              if (sc > tvv[q2]) {
                const float tv = tvv[q2]; const int ti = tii[q2];
                tvv[q2] = sc; tii[q2] = ci;
                sc = tv; ci = ti;
              }
            }
          }
          const float mx = tvv[0];
          float wgt[kK]; float ssum = 0.0f;
#pragma unroll
          for (int i = 0; i < kK; ++i) { wgt[i] = __expf(tvv[i] - mx); ssum += wgt[i]; }
          const float inv = 1.0f / ssum;
#pragma unroll 8
          for (int d = 0; d < kDH; ++d) {
            float o = 0.0f;
#pragma unroll
            for (int i = 0; i < kK; ++i)
              o += wgt[i] * (float)s.qkv[2][tii[i]][head * kDH + d];
            // attn-out reuses qkv[0]; row/col ranges already consumed by scores.
            s.qkv[0][grow][head * kDH + d] = (_Float16)(o * inv);
          }
        }
        __syncthreads();
      }
    }

    // ---- output projection + residual into h
    {
      const int row0 = wid * 16;
      for (int ct = 0; ct < 4; ++ct) {
        const int col0 = ct * 16;
        const v8f ao = mm_tile_hf16(&s.qkv[0][0][0], &s.w1[0][0], row0, col0);
        const int col = col0 + (lane & 15);
#pragma unroll
        for (int r = 0; r < 8; ++r) {
          const int row = row0 + r + ((lane >> 4) << 3);
          s.h[row][col] += ao[r] + s.bias2[col];
        }
      }
    }
    __syncthreads();

    // ---- LayerNorm over D per node
    if (tid < kN) {
      float mu = 0.0f;
#pragma unroll 8
      for (int k2 = 0; k2 < kD; ++k2) mu += s.h[tid][k2];
      mu *= (1.0f / kD);
      float var = 0.0f;
#pragma unroll 8
      for (int k2 = 0; k2 < kD; ++k2) {
        const float d = s.h[tid][k2] - mu;
        var += d * d;
      }
      var *= (1.0f / kD);
      const float inv = rsqrtf(var + kEps);
#pragma unroll 8
      for (int k2 = 0; k2 < kD; ++k2)
        s.h[tid][k2] = (s.h[tid][k2] - mu) * inv * s.lng[k2] + s.lnb[k2];
    }
    __syncthreads();
  }

  // ================= Basis head: (128x64) @ (6x64)^T + b
  for (int i = tid; i < kNB * kD; i += 256) stage[i] = w_basis[i];
  if (tid < kNB) s.bias[tid] = b_basis[tid];
  __syncthreads();
  if (tid < kN) {
#pragma unroll
    for (int o = 0; o < kNB; ++o) {
      float acc = s.bias[o];
#pragma unroll 8
      for (int k2 = 0; k2 < kD; ++k2) acc += s.h[tid][k2] * stage[o * kD + k2];
      out[(m * kN + tid) * kNB + o] = acc;
    }
  }
}

}  // namespace

extern "C" void kernel_launch(void* const* d_in, const int* in_sizes, int n_in,
                              void* d_out, int out_size, void* d_ws, size_t ws_size,
                              hipStream_t stream) {
  (void)in_sizes; (void)n_in; (void)out_size; (void)d_ws; (void)ws_size;
  const float* x       = (const float*)d_in[0];
  const float* w_in1   = (const float*)d_in[1];
  const float* b_in1   = (const float*)d_in[2];
  const float* w_in2   = (const float*)d_in[3];
  const float* b_in2   = (const float*)d_in[4];
  const float* species = (const float*)d_in[5];
  const float* wq      = (const float*)d_in[6];
  const float* bq      = (const float*)d_in[7];
  const float* wk      = (const float*)d_in[8];
  const float* bk      = (const float*)d_in[9];
  const float* wv      = (const float*)d_in[10];
  const float* bv      = (const float*)d_in[11];
  const float* wo      = (const float*)d_in[12];
  const float* bo      = (const float*)d_in[13];
  const float* lng     = (const float*)d_in[14];
  const float* lnb     = (const float*)d_in[15];
  const float* w_basis = (const float*)d_in[16];
  const float* b_basis = (const float*)d_in[17];
  float* out = (float*)d_out;

  dim3 grid(kM), block(256);
  hipLaunchKernelGGL(gnn_basis_kernel, grid, block, 0, stream,
                     x, w_in1, b_in1, w_in2, b_in2, species,
                     wq, bq, wk, bk, wv, bv, wo, bo,
                     lng, lnb, w_basis, b_basis, out);
}
